// MyModel_61933428415788
// MI455X (gfx1250) — matrix operand order repaired and checked by reading, not compiled, not run on hardware
//
#include <hip/hip_runtime.h>
#include <stdint.h>

// ---------------------------------------------------------------------------
// LSTM on MI455X (gfx1250), wave32 + WMMA.
//
// gates[16x80] = A[16x32] @ W[32x80],  A = [x_t(8) | h(20) | 0(4)] per 16-batch
// tile, computed with 5x v_wmma_f32_16x16x32_f16 per step (one N-tile per
// wave of a 5-wave workgroup). Weight fragments live in VGPRs for the whole
// T loop. x is streamed via global_load_async_to_lds_b128 into a ring buffer
// (ASYNCcnt-tracked, depth 7). Activations use native v_tanh_f32 when the
// builtin exists (sigmoid(x) = 0.5*tanh(0.5x)+0.5 -> 1 trans op each).
// ---------------------------------------------------------------------------

typedef _Float16 v16h __attribute__((ext_vector_type(16)));
typedef _Float16 v4h  __attribute__((ext_vector_type(4)));
typedef float    v8f  __attribute__((ext_vector_type(8)));
typedef float    v4f  __attribute__((ext_vector_type(4)));

namespace {
constexpr int kT    = 2048;
constexpr int kB    = 4096;
constexpr int kIn   = 8;
constexpr int kH    = 20;
constexpr int kG    = 80;   // 4*kH
constexpr int kK    = 32;   // kIn + kH padded to 32
constexpr int kRows = 16;   // batch rows per workgroup (WMMA M)
constexpr int kWG   = 160;  // 5 waves: one per 16-col gate tile
constexpr int kPF   = 6;    // async prefetch depth
constexpr int kRing = 8;    // x ring slots (power of 2)
constexpr int kXBlk = kRows * kIn;  // 128 floats of x per step per group
}

// ISA 7.12.2: 16-bit A 16x32 — lane = m + 16*(hi&1), half = 8*(hi>>1)+(k&7),
// hi = k>>3.  (lanes 0-15: K 0-7 then 16-23; lanes 16-31: K 8-15 then 24-31)
__device__ __forceinline__ int a_slot(int m, int k) {
  int hi = k >> 3;
  return (m + ((hi & 1) << 4)) * 16 + ((hi >> 1) << 3) + (k & 7);
}
// 16-bit B 32x16: lanes 0-15 hold K 0-7 then 16-23, lanes 16-31 hold K 8-15
// then 24-31, N = lane%16, half = 8*(k>>4)+(k&7).
__device__ __forceinline__ int b_slot(int k, int n) {
  return (n >> 4) * 512 + ((n & 15) + (((k >> 3) & 1) << 4)) * 16 + ((k >> 4) << 3) + (k & 7);
}

#if __has_builtin(__builtin_amdgcn_tanhf)
__device__ __forceinline__ float tanh_f(float x) {
  return __builtin_amdgcn_tanhf(x);
}
__device__ __forceinline__ float sigm(float x) {
  return __builtin_fmaf(__builtin_amdgcn_tanhf(0.5f * x), 0.5f, 0.5f);
}
#else
__device__ __forceinline__ float sigm(float x) {
  return __builtin_amdgcn_rcpf(1.0f + __builtin_amdgcn_exp2f(-1.44269504f * x));
}
__device__ __forceinline__ float tanh_f(float x) {
  return 2.0f * __builtin_amdgcn_rcpf(1.0f + __builtin_amdgcn_exp2f(-2.88539008f * x)) - 1.0f;
}
#endif

// Async copy 16B/lane from global to LDS ring slot (ASYNCcnt-tracked).
__device__ __forceinline__ void issue_x(const float* x, int t, int b0, int lane,
                                        uint32_t ring_base) {
  uint32_t lds = ring_base + (uint32_t)((t & (kRing - 1)) * kXBlk * 4 + lane * 16);
  uint64_t ga  = (uint64_t)(uintptr_t)(x + ((size_t)t * kB + b0) * kIn)
               + (uint64_t)(lane * 16);
  asm volatile("global_load_async_to_lds_b128 %0, %1, off"
               :: "v"(lds), "v"(ga) : "memory");
}

// Wave0: convert this step's 128 raw x floats into f16 A-fragment slots.
__device__ __forceinline__ void stage_x(const float* xr, _Float16* as, int lane) {
  v4f xv = *(const v4f*)(xr + lane * 4);                 // floats 4l..4l+3
  v4h hv = __builtin_convertvector(xv, v4h);
  // row m = lane/2, features i0..i0+3 (A slots for k<8: lane m, halves k)
  *(v4h*)(as + (lane >> 1) * 16 + ((lane & 1) << 2)) = hv;
}

__global__ __launch_bounds__(kWG) void lstm_wmma(
    const float* __restrict__ x,    const float* __restrict__ W_ih,
    const float* __restrict__ b_ih, const float* __restrict__ W_hh,
    const float* __restrict__ b_hh, const float* __restrict__ hx0,
    const float* __restrict__ cx0,  float* __restrict__ out) {
  __shared__ __align__(32) _Float16 a_stage[kRows * kK];      // 1 KB
  __shared__ __align__(32) _Float16 w_stage[5 * 512];         // 5 KB
  __shared__ __align__(16) float    gates_T[kG * kRows];      // 5 KB, [n][m]
  __shared__ __align__(16) float    xring[kRing * kXBlk];     // 4 KB

  const int tid  = threadIdx.x;
  const int lane = tid & 31;
  const int wv   = tid >> 5;           // 0..4 = gate N-tile
  const int b0   = blockIdx.x * kRows; // batch base row

  // ---- init A staging: h0 into h-slots, zeros into x-slots and K-pad ----
  for (int idx = tid; idx < kRows * kK; idx += kWG) {
    int m = idx & 15, k = idx >> 4;
    float v = (k >= kIn && k < kIn + kH) ? hx0[k - kIn] : 0.0f;
    a_stage[a_slot(m, k)] = (_Float16)v;
  }
  // ---- init weight fragments: W = [W_ih ; W_hh ; 0], f16, B layout ----
  for (int idx = tid; idx < kK * kG; idx += kWG) {
    int k = idx / kG, n = idx % kG;
    float v = 0.0f;
    if (k < kIn)            v = W_ih[k * kG + n];
    else if (k < kIn + kH)  v = W_hh[(k - kIn) * kG + n];
    w_stage[b_slot(k, n)] = (_Float16)v;
  }

  const uint32_t ring_base = (uint32_t)(uintptr_t)(&xring[0]);
  if (tid < 32) {                       // prologue: prefetch x for t=0..kPF
    for (int t = 0; t <= kPF; ++t) issue_x(x, t, b0, lane, ring_base);
  }

  __syncthreads();                      // publish w_stage / a_stage init

  // ---- per-lane persistent state ----
  v16h bfrag = *(const v16h*)&w_stage[wv * 512 + lane * 16];  // pinned B tile
  const int   ncol = wv * 16 + (lane & 15);                   // gate column
  const float bias = b_ih[ncol] + b_hh[ncol];
  const int   mrow = (lane >> 4) * 8;                         // D-tile row base

  // elementwise ownership: 2 cells (m,j) per lane, 320 cells total
  const int cell0 = 2 * tid, cell1 = 2 * tid + 1;
  const int m0c = cell0 / kH, j0c = cell0 % kH;
  const int m1c = cell1 / kH, j1c = cell1 % kH;
  float c0 = cx0[j0c], c1 = cx0[j1c];
  float hy0 = 0.0f, hy1 = 0.0f;
  const int gi0 = j0c * kRows + m0c, gi1 = j1c * kRows + m1c;
  const int as0 = a_slot(m0c, kIn + j0c), as1 = a_slot(m1c, kIn + j1c);

  if (tid < 32) {                       // stage x(0)
    asm volatile("s_wait_asynccnt 6" ::: "memory");  // oldest (t=0) landed
    stage_x(&xring[0], a_stage, lane);
  }

  for (int t = 0; t < kT; ++t) {
    __syncthreads();  // x(t) + h(t) staged in a_stage, visible to all waves

    // ---- WMMA: D = A(16x32,f16) x B(32x16,f16) + bias ----
    v16h afrag = *(const v16h*)&a_stage[lane * 16];
    v8f  cacc;
#pragma unroll
    for (int r = 0; r < 8; ++r) cacc[r] = bias;
    v8f d = __builtin_amdgcn_wmma_f32_16x16x32_f16(
        false, afrag, false, bfrag, (short)0, cacc, false, false);

    // dump D transposed: gates_T[n][m0..m0+7]
    float* gp = &gates_T[ncol * kRows + mrow];
    *(v4f*)gp       = d.lo;
    *(v4f*)(gp + 4) = d.hi;

    __syncthreads();  // gates visible; all A-fragment reads complete

    // ---- elementwise LSTM cell, 2 (m,j) cells per lane ----
    {
      float ig = sigm(gates_T[gi0]);
      float fg = sigm(gates_T[gi0 + kH * kRows]);
      float gg = tanh_f(gates_T[gi0 + 2 * kH * kRows]);
      float og = sigm(gates_T[gi0 + 3 * kH * kRows]);
      c0  = fg * c0 + ig * gg;
      hy0 = og * tanh_f(c0);
      a_stage[as0] = (_Float16)hy0;     // h(t+1) into A fragment
    }
    {
      float ig = sigm(gates_T[gi1]);
      float fg = sigm(gates_T[gi1 + kH * kRows]);
      float gg = tanh_f(gates_T[gi1 + 2 * kH * kRows]);
      float og = sigm(gates_T[gi1 + 3 * kH * kRows]);
      c1  = fg * c1 + ig * gg;
      hy1 = og * tanh_f(c1);
      a_stage[as1] = (_Float16)hy1;
    }

    // ---- stream next x block (wave 0 only; disjoint a_stage slots) ----
    if (t + 1 < kT) {
      if (tid < 32) {
        if (t + kPF + 1 < kT) {
          asm volatile("s_wait_asynccnt 5" ::: "memory");  // (t+1) landed
        } else {
          asm volatile("s_wait_asynccnt 0" ::: "memory");  // drain at tail
        }
        stage_x(&xring[((t + 1) & (kRing - 1)) * kXBlk], a_stage, lane);
        if (t + kPF + 1 < kT) issue_x(x, t + kPF + 1, b0, lane, ring_base);
      }
    }
  }

  // ---- final hidden state: out[b][j] ----
  out[(b0 + m0c) * kH + j0c] = hy0;
  out[(b0 + m1c) * kH + j1c] = hy1;
}

extern "C" void kernel_launch(void* const* d_in, const int* in_sizes, int n_in,
                              void* d_out, int out_size, void* d_ws,
                              size_t ws_size, hipStream_t stream) {
  (void)in_sizes; (void)n_in; (void)out_size; (void)d_ws; (void)ws_size;
  const float* x    = (const float*)d_in[0];
  const float* W_ih = (const float*)d_in[1];
  const float* b_ih = (const float*)d_in[2];
  const float* W_hh = (const float*)d_in[3];
  const float* b_hh = (const float*)d_in[4];
  const float* hx0  = (const float*)d_in[5];
  const float* cx0  = (const float*)d_in[6];
  float* out = (float*)d_out;

  dim3 grid(kB / kRows);   // 256 workgroups, one per 16-batch tile
  dim3 block(kWG);         // 5 waves
  hipLaunchKernelGGL(lstm_wmma, grid, block, 0, stream,
                     x, W_ih, b_ih, W_hh, b_hh, hx0, cx0, out);
}
